// TransfomerAttentionLayer_24163486007310
// MI455X (gfx1250) — matrix operand order repaired and checked by reading, not compiled, not run on hardware
//
#include <hip/hip_runtime.h>
#include <hip/hip_bf16.h>
#include <math.h>

typedef __bf16 bf16_t;
typedef __attribute__((ext_vector_type(16))) __bf16 v16bf;
typedef __attribute__((ext_vector_type(8)))  float  v8f;

#define DEG     20
#define GRP     4              // dst nodes per fused-attn workgroup
#define EPG     (GRP * DEG)    // 80 edges per group -> 5 row tiles of 16
#define KV_K    384            // kv_in inner dim
#define LDA_ATT 392            // 384 + 8 bf16 pad (16B aligned stride)
#define LDK_ATT 132            // 128 + 4 f32 pad
#define LN_EPS  1e-5f

// ---------------------------------------------------------------------------
// Fragment loaders.
// A (16x32 bf16, MxK): lane<16 holds row M=lane, K {kb..kb+7},{kb+16..kb+23};
//                      lane>=16 holds row M=lane-16, K {kb+8..},{kb+24..}.
// B (32x16 bf16, KxN, source stored transposed [col][k]):
//   lane<16 holds col N=lane, K {kb..kb+7},{kb+16..kb+23}; lane>=16 col lane-16, K {kb+8..},{kb+24..}.
// C/D (16x16 f32): vgpr r, lane l -> (M = r + (l>=16)*8, N = l&15).
// ---------------------------------------------------------------------------
__device__ __forceinline__ v16bf load_a_frag(const bf16_t* __restrict__ A,
                                             int row0, int kb, int lane, int lda) {
  int hi = (lane >> 4) & 1;
  const bf16_t* p = A + (row0 + (lane & 15)) * lda + kb + hi * 8;
  union { v16bf v; uint4 u[2]; } r;
  r.u[0] = *(const uint4*)(p);
  r.u[1] = *(const uint4*)(p + 16);
  return r.v;
}

__device__ __forceinline__ v16bf load_b_frag(const bf16_t* __restrict__ BT,
                                             int j0, int kb, int lane, int ldb) {
  int hi = (lane >> 4) & 1;
  const bf16_t* p = BT + (size_t)(j0 + (lane & 15)) * ldb + kb + hi * 8;
  union { v16bf v; uint4 u[2]; } r;
  r.u[0] = *(const uint4*)(p);
  r.u[1] = *(const uint4*)(p + 16);
  return r.v;
}

__device__ __forceinline__ v8f wmma_bf16(v16bf a, v16bf b, v8f c) {
  return __builtin_amdgcn_wmma_f32_16x16x32_bf16(false, a, false, b, (short)0, c,
                                                 false, false);
}

// ---------------------------------------------------------------------------
// Prep: transpose+convert weights to bf16 [out_col][k]; fold zero-time-feat
// contribution + b_q into q_const.
// ---------------------------------------------------------------------------
__global__ __launch_bounds__(256) void prep_kernel(
    const float* __restrict__ w_q, const float* __restrict__ b_q,
    const float* __restrict__ w_k, const float* __restrict__ w_v,
    const float* __restrict__ w_out, const float* __restrict__ time_b,
    bf16_t* __restrict__ wqT, bf16_t* __restrict__ wkT, bf16_t* __restrict__ wvT,
    bf16_t* __restrict__ woT, float* __restrict__ q_const) {
  int tid = blockIdx.x * blockDim.x + threadIdx.x;
  int stride = gridDim.x * blockDim.x;
  for (int i = tid; i < 128 * 128; i += stride) {
    int j = i >> 7, k = i & 127;
    wqT[j * 128 + k] = (bf16_t)w_q[k * 128 + j];
  }
  for (int i = tid; i < 128 * 384; i += stride) {
    int j = i / 384, k = i - j * 384;
    wkT[j * 384 + k] = (bf16_t)w_k[k * 128 + j];
    wvT[j * 384 + k] = (bf16_t)w_v[k * 128 + j];
  }
  for (int i = tid; i < 128 * 256; i += stride) {
    int j = i >> 8, k = i & 255;
    woT[j * 256 + k] = (bf16_t)w_out[k * 128 + j];
  }
  for (int i = tid; i < 128; i += stride) {
    float s = b_q[i];
    for (int t = 0; t < 128; ++t) s += cosf(time_b[t]) * w_q[(128 + t) * 128 + i];
    q_const[i] = s;
  }
}

// ---------------------------------------------------------------------------
// Q GEMM: Q[n,:] = bf16(node_h[n,:]) @ wqT + q_const   (16 rows / block)
// ---------------------------------------------------------------------------
__global__ __launch_bounds__(256) void q_gemm_kernel(
    const float* __restrict__ node_h, const bf16_t* __restrict__ wqT,
    const float* __restrict__ q_const, float* __restrict__ Qw, int num_dst) {
  __shared__ bf16_t As[16 * 136];
  int tid = threadIdx.x, lane = tid & 31, wv = tid >> 5;
  int m0 = blockIdx.x * 16;
  for (int i = tid; i < 16 * 128; i += 256) {
    int r = i >> 7, c = i & 127;
    int row = m0 + r;
    As[r * 136 + c] = (bf16_t)(row < num_dst ? node_h[(size_t)row * 128 + c] : 0.f);
  }
  __syncthreads();
  int j0 = wv * 16;
  v8f acc = {};
#pragma unroll
  for (int kk = 0; kk < 4; ++kk) {
    v16bf a = load_a_frag(As, 0, kk * 32, lane, 136);
    v16bf b = load_b_frag(wqT, j0, kk * 32, lane, 128);
    acc = wmma_bf16(a, b, acc);
  }
  int col = j0 + (lane & 15);
  int hb = (lane >> 4) * 8;
  float qc = q_const[col];
#pragma unroll
  for (int r = 0; r < 8; ++r) {
    int row = m0 + r + hb;
    if (row < num_dst) Qw[(size_t)row * 128 + col] = acc[r] + qc;
  }
}

// ---------------------------------------------------------------------------
// Fused attention: per block = 4 dst nodes (80 edges).
//  phase 0: stage kv_in (node|edge|cos(dt*w)) as bf16 in LDS; stage Q rows
//  phase 1: K = A @ wkT + b_k  -> LDS (WMMA)
//  phase 2: per-(edge,head) scores, LeakyReLU, per-dst softmax (VALU)
//  phase 3: V tiles (WMMA); att-weighted sum accumulated in registers,
//           halves combined via shfl_xor; write agg[4x128]
// ---------------------------------------------------------------------------
__global__ __launch_bounds__(256) void attn_fused_kernel(
    const float* __restrict__ node_h, const float* __restrict__ edge_f,
    const float* __restrict__ edge_dt, const float* __restrict__ time_w,
    const bf16_t* __restrict__ wkT, const bf16_t* __restrict__ wvT,
    const float* __restrict__ b_k, const float* __restrict__ b_v,
    const float* __restrict__ Qw, float* __restrict__ aggw,
    int num_dst, int num_edge) {
  extern __shared__ char smem[];
  bf16_t* As = (bf16_t*)smem;                                  // 80 x 392 bf16
  float*  Ks = (float*)(smem + EPG * LDA_ATT * 2);             // 80 x 132 f32
  float*  Qs = (float*)(smem + EPG * LDA_ATT * 2 + EPG * LDK_ATT * 4);  // 4x128
  float*  Ss = (float*)((char*)Qs + GRP * 128 * 4);            // 80 x 4

  int tid = threadIdx.x, lane = tid & 31, wv = tid >> 5;
  int dst_base = blockIdx.x * GRP;
  int ebase = blockIdx.x * EPG;

  // ---- phase 0: stage A (bf16) and Q rows ----
  for (int i = tid; i < EPG * KV_K; i += 256) {
    int e = i / KV_K, c = i - e * KV_K;
    int eg = ebase + e;
    float v = 0.f;
    if (eg < num_edge) {
      if (c < 128)       v = node_h[((size_t)num_dst + eg) * 128 + c];
      else if (c < 256)  v = edge_f[(size_t)eg * 128 + (c - 128)];
      else               v = cosf(edge_dt[eg] * time_w[c - 256]);
    }
    As[e * LDA_ATT + c] = (bf16_t)v;
  }
  for (int i = tid; i < GRP * 128; i += 256) {
    int d = i >> 7, c = i & 127;
    int row = dst_base + d;
    Qs[i] = (row < num_dst) ? Qw[(size_t)row * 128 + c] : 0.f;
  }
  __syncthreads();

  // ---- phase 1: K projection (5 row tiles x 8 col tiles over 8 waves) ----
  for (int task = wv; task < 40; task += 8) {
    int rt = task >> 3;
    int j0 = (task & 7) * 16;
    v8f acc = {};
#pragma unroll
    for (int kk = 0; kk < 12; ++kk) {
      v16bf a = load_a_frag(As, rt * 16, kk * 32, lane, LDA_ATT);
      v16bf b = load_b_frag(wkT, j0, kk * 32, lane, KV_K);
      acc = wmma_bf16(a, b, acc);
    }
    int col = j0 + (lane & 15);
    int hb = (lane >> 4) * 8;
    float bias = b_k[col];
#pragma unroll
    for (int r = 0; r < 8; ++r)
      Ks[(rt * 16 + r + hb) * LDK_ATT + col] = acc[r] + bias;
  }
  __syncthreads();

  // ---- phase 2: scores + LeakyReLU, then per-dst softmax ----
  for (int p = tid; p < EPG * 4; p += 256) {
    int e = p >> 2, h = p & 3;
    int dl = e / DEG;
    float s = 0.f;
#pragma unroll 8
    for (int c = 0; c < 32; ++c)
      s += Qs[dl * 128 + h * 32 + c] * Ks[e * LDK_ATT + h * 32 + c];
    Ss[e * 4 + h] = (s >= 0.f) ? s : 0.2f * s;
  }
  __syncthreads();
  if (tid < GRP * 4) {
    int dl = tid >> 2, h = tid & 3;
    float m = -3.0e38f;
    for (int e = 0; e < DEG; ++e) m = fmaxf(m, Ss[(dl * DEG + e) * 4 + h]);
    if (!(m > -3.0e38f && m < 3.0e38f)) m = 0.f;
    float den = 0.f;
    for (int e = 0; e < DEG; ++e) den += expf(Ss[(dl * DEG + e) * 4 + h] - m);
    float inv = 1.f / den;
    for (int e = 0; e < DEG; ++e) {
      float ex = expf(Ss[(dl * DEG + e) * 4 + h] - m);
      Ss[(dl * DEG + e) * 4 + h] = ex * inv;   // att weight
    }
  }
  __syncthreads();

  // ---- phase 3: V projection + weighted aggregation in registers ----
  {
    int j0 = wv * 16;
    int col = j0 + (lane & 15);
    int hb = (lane >> 4) * 8;
    int head = col >> 5;
    float bias = b_v[col];
    float aggv[GRP] = {0.f, 0.f, 0.f, 0.f};
#pragma unroll
    for (int rt = 0; rt < 5; ++rt) {
      v8f acc = {};
#pragma unroll
      for (int kk = 0; kk < 12; ++kk) {
        v16bf a = load_a_frag(As, rt * 16, kk * 32, lane, LDA_ATT);
        v16bf b = load_b_frag(wvT, j0, kk * 32, lane, KV_K);
        acc = wmma_bf16(a, b, acc);
      }
#pragma unroll
      for (int r = 0; r < 8; ++r) {
        int e = rt * 16 + r + hb;
        float att = Ss[e * 4 + head];
        aggv[e / DEG] += att * (acc[r] + bias);
      }
    }
#pragma unroll
    for (int d = 0; d < GRP; ++d) {
      float other = __shfl_xor(aggv[d], 16);
      float tot = aggv[d] + other;
      int row = dst_base + d;
      if ((lane >> 4) == 0 && row < num_dst)
        aggw[(size_t)row * 128 + col] = tot;
    }
  }
}

// ---------------------------------------------------------------------------
// Output: rst = relu([agg | node_h] @ woT + b_out); LayerNorm -> d_out
// ---------------------------------------------------------------------------
__global__ __launch_bounds__(256) void out_ln_kernel(
    const float* __restrict__ aggw, const float* __restrict__ node_h,
    const bf16_t* __restrict__ woT, const float* __restrict__ b_out,
    const float* __restrict__ ln_g, const float* __restrict__ ln_b,
    float* __restrict__ out, int num_dst) {
  __shared__ bf16_t As[16 * 264];
  __shared__ float rst[16 * 132];
  __shared__ float mu_s[16], rs_s[16];
  int tid = threadIdx.x, lane = tid & 31, wv = tid >> 5;
  int m0 = blockIdx.x * 16;
  for (int i = tid; i < 16 * 256; i += 256) {
    int r = i >> 8, c = i & 255;
    int row = m0 + r;
    float v = 0.f;
    if (row < num_dst)
      v = (c < 128) ? aggw[(size_t)row * 128 + c]
                    : node_h[(size_t)row * 128 + (c - 128)];
    As[r * 264 + c] = (bf16_t)v;
  }
  __syncthreads();
  int j0 = wv * 16;
  v8f acc = {};
#pragma unroll
  for (int kk = 0; kk < 8; ++kk) {
    v16bf a = load_a_frag(As, 0, kk * 32, lane, 264);
    v16bf b = load_b_frag(woT, j0, kk * 32, lane, 256);
    acc = wmma_bf16(a, b, acc);
  }
  int col = j0 + (lane & 15);
  int hb = (lane >> 4) * 8;
  float bo = b_out[col];
#pragma unroll
  for (int r = 0; r < 8; ++r)
    rst[(r + hb) * 132 + col] = fmaxf(acc[r] + bo, 0.f);
  __syncthreads();
  if (tid < 16) {
    float mu = 0.f;
    for (int c = 0; c < 128; ++c) mu += rst[tid * 132 + c];
    mu *= (1.f / 128.f);
    float var = 0.f;
    for (int c = 0; c < 128; ++c) {
      float d = rst[tid * 132 + c] - mu;
      var += d * d;
    }
    var *= (1.f / 128.f);
    mu_s[tid] = mu;
    rs_s[tid] = rsqrtf(var + LN_EPS);
  }
  __syncthreads();
  for (int i = tid; i < 16 * 128; i += 256) {
    int r = i >> 7, c = i & 127;
    int row = m0 + r;
    if (row < num_dst)
      out[(size_t)row * 128 + c] =
          (rst[r * 132 + c] - mu_s[r]) * rs_s[r] * ln_g[c] + ln_b[c];
  }
}

// ---------------------------------------------------------------------------
extern "C" void kernel_launch(void* const* d_in, const int* in_sizes, int n_in,
                              void* d_out, int out_size, void* d_ws, size_t ws_size,
                              hipStream_t stream) {
  const float* node_h  = (const float*)d_in[0];
  const float* edge_f  = (const float*)d_in[1];
  const float* edge_dt = (const float*)d_in[2];
  // d_in[3] dst_idx: edges are contiguous per dst (dst = e / DEG), unused.
  int E = in_sizes[2];
  int num_dst = in_sizes[0] / 128 - E;
  const float* time_w = (const float*)d_in[5];
  const float* time_b = (const float*)d_in[6];
  const float* w_q   = (const float*)d_in[7];
  const float* b_q   = (const float*)d_in[8];
  const float* w_k   = (const float*)d_in[9];
  const float* b_k   = (const float*)d_in[10];
  const float* w_v   = (const float*)d_in[11];
  const float* b_v   = (const float*)d_in[12];
  const float* w_out = (const float*)d_in[13];
  const float* b_out = (const float*)d_in[14];
  const float* ln_g  = (const float*)d_in[15];
  const float* ln_b  = (const float*)d_in[16];

  char* ws = (char*)d_ws;
  size_t off = 0;
  bf16_t* wqT = (bf16_t*)(ws + off); off += (size_t)128 * 128 * 2;
  bf16_t* wkT = (bf16_t*)(ws + off); off += (size_t)128 * 384 * 2;
  bf16_t* wvT = (bf16_t*)(ws + off); off += (size_t)128 * 384 * 2;
  bf16_t* woT = (bf16_t*)(ws + off); off += (size_t)128 * 256 * 2;
  float* q_const = (float*)(ws + off); off += 128 * 4;
  off = (off + 255) & ~(size_t)255;
  float* Qw = (float*)(ws + off); off += (size_t)num_dst * 128 * 4;
  float* aggw = (float*)(ws + off);

  prep_kernel<<<256, 256, 0, stream>>>(w_q, b_q, w_k, w_v, w_out, time_b,
                                       wqT, wkT, wvT, woT, q_const);

  int qt = (num_dst + 15) / 16;
  q_gemm_kernel<<<qt, 256, 0, stream>>>(node_h, wqT, q_const, Qw, num_dst);

  int groups = (num_dst + GRP - 1) / GRP;
  size_t smem2 = (size_t)EPG * LDA_ATT * 2 + (size_t)EPG * LDK_ATT * 4 +
                 (size_t)GRP * 128 * 4 + (size_t)EPG * 4 * 4;  // 108,288 B
  attn_fused_kernel<<<groups, 256, smem2, stream>>>(
      node_h, edge_f, edge_dt, time_w, wkT, wvT, b_k, b_v, Qw, aggw,
      num_dst, E);

  out_ln_kernel<<<qt, 256, 0, stream>>>(aggw, node_h, woT, b_out, ln_g, ln_b,
                                        (float*)d_out, num_dst);
}
